// tang_vit_13941463843466
// MI455X (gfx1250) — matrix operand order repaired and checked by reading, not compiled, not run on hardware
//
#include <hip/hip_runtime.h>
#include <math.h>

// ---------- types ----------
typedef __attribute__((ext_vector_type(16))) __bf16 v16bf;
typedef __attribute__((ext_vector_type(8)))  float  v8f;

union Frag {
    v16bf v;
    unsigned short u[16];
    uint4 q[2];
};

__device__ inline unsigned short f2bf(float f) {
    unsigned u = __builtin_bit_cast(unsigned, f);
    u += 0x7fffu + ((u >> 16) & 1u);
    return (unsigned short)(u >> 16);
}

__device__ inline v8f vzero() {
    v8f z;
    #pragma unroll
    for (int i = 0; i < 8; ++i) z[i] = 0.f;
    return z;
}

__device__ inline v8f wmma_bf(const Frag& a, const Frag& b, v8f c) {
    // D = A(16x32 bf16) * B(32x16 bf16) + C(f32)
    return __builtin_amdgcn_wmma_f32_16x16x32_bf16(false, a.v, false, b.v,
                                                   (short)0, c, false, false);
}

__device__ inline float wred_sum(float x) {
    #pragma unroll
    for (int m = 16; m >= 1; m >>= 1) x += __shfl_xor(x, m, 32);
    return x;
}
__device__ inline float wred_max(float x) {
    #pragma unroll
    for (int m = 16; m >= 1; m >>= 1) x = fmaxf(x, __shfl_xor(x, m, 32));
    return x;
}

// ---------- model dims ----------
#define BATCH 32
#define SEQ   197
#define DIM   768
#define NHEAD 12
#define DH    64
#define NLAY  12
#define MROWS (BATCH * SEQ)      // 6304
#define MLPD  3072
#define ATTP  224                 // padded K for att@v (7*32)

// ---------- tile params ----------
#define BM 128
#define BN 64
#define BK 32
#define ASTR 40
#define BSTR 40

// ======================================================================
// cls token + pos row 0
// ======================================================================
__global__ void k_cls_pos(float* __restrict__ X, const float* __restrict__ cls,
                          const float* __restrict__ pos) {
    int idx = blockIdx.x * 256 + threadIdx.x;       // 0 .. 32*768-1
    int n = idx / DIM, d = idx % DIM;
    X[(size_t)n * SEQ * DIM + d] = cls[d] + pos[d];
}

// ======================================================================
// big tiled GEMM, bf16 WMMA, fp32 accumulate.
// MODE 0: embed  (A = patchify(images) gather, epilogue = +bias +pos -> X rows 1..196)
// MODE 1: mlp1   (A = bf16 ln buffer, epilogue = GELU -> bf16)
// MODE 2: mlp2   (A = bf16 h1 buffer, epilogue = +bias, X += )
// ======================================================================
template<int MODE>
__global__ void __launch_bounds__(256)
k_gemm(const unsigned short* __restrict__ Abf,
       const float* __restrict__ img,
       const float* __restrict__ W,       // [K][N] row-major fp32
       const float* __restrict__ bias,    // [N]
       const float* __restrict__ pos,
       float* __restrict__ Xout,
       unsigned short* __restrict__ Obf,
       int M, int N, int K)
{
    __shared__ unsigned short As[BM * ASTR];
    __shared__ unsigned short Bs[BN * BSTR];

    const int t = threadIdx.x;
    const int lane = t & 31;
    const int wv = t >> 5;
    const int wy = wv >> 1, wx = wv & 1;
    const int bm = blockIdx.x, bn = blockIdx.y;

    v8f acc[2][2];
    acc[0][0] = vzero(); acc[0][1] = vzero();
    acc[1][0] = vzero(); acc[1][1] = vzero();

    const int arow = t >> 1;            // 0..127
    const int asub = (t & 1) * 16;      // 0/16
    const int brow = t >> 3;            // 0..31  (k within tile)
    const int bn0  = (t & 7) * 8;       // 0..56  (n within tile)

    const int lm   = lane & 15;
    const int koff = (lane >> 4) * 8;
    const int kb   = (lane >> 4) * 16;

    for (int kt = 0; kt < K; kt += BK) {
        __syncthreads();
        // ---- stage A tile (BM x BK) as bf16 ----
        {
            int gm = bm * BM + arow;
            unsigned short* ad = &As[arow * ASTR + asub];
            if (MODE == 0) {
                // patchify gather: row gm -> (n, p); col k -> (c, ph, pw)
                int nimg = gm / 196, p = gm % 196;
                int py = p / 14, px = p % 14;
                int kg = kt + asub;                 // multiple of 16 -> pw runs 0..15
                int c = kg >> 8, ph = (kg >> 4) & 15;
                const float4* s4 = (const float4*)(img +
                    (((size_t)(nimg * 3 + c) * 224) + py * 16 + ph) * 224 + px * 16);
                #pragma unroll
                for (int i = 0; i < 4; ++i) {
                    float4 f = s4[i];
                    ad[i*4+0] = f2bf(f.x); ad[i*4+1] = f2bf(f.y);
                    ad[i*4+2] = f2bf(f.z); ad[i*4+3] = f2bf(f.w);
                }
            } else {
                if (gm < M) {
                    const uint4* s = (const uint4*)(Abf + (size_t)gm * K + kt + asub);
                    *(uint4*)(ad)     = s[0];
                    *(uint4*)(ad + 8) = s[1];
                } else {
                    #pragma unroll
                    for (int i = 0; i < 16; ++i) ad[i] = 0;
                }
            }
        }
        // ---- stage B tile (BK x BN) fp32 -> bf16, transposed Bs[n][k] ----
        {
            const float* wp = W + (size_t)(kt + brow) * N + bn * BN + bn0;
            const float4* s4 = (const float4*)wp;
            float4 f0 = s4[0], f1 = s4[1];
            float ff[8] = {f0.x, f0.y, f0.z, f0.w, f1.x, f1.y, f1.z, f1.w};
            #pragma unroll
            for (int i = 0; i < 8; ++i) Bs[(bn0 + i) * BSTR + brow] = f2bf(ff[i]);
            if (kt + BK < K)  // hint next weight tile (global_prefetch_b8)
                __builtin_prefetch(wp + (size_t)BK * N, 0, 1);
        }
        __syncthreads();

        // ---- wave compute: 32x32 per wave via 4 WMMAs ----
        Frag a0, a1, b0, b1;
        int r0 = wy * 32 + lm;
        a0.q[0] = *(const uint4*)&As[r0 * ASTR + kb];
        a0.q[1] = *(const uint4*)&As[r0 * ASTR + kb + 8];
        a1.q[0] = *(const uint4*)&As[(r0 + 16) * ASTR + kb];
        a1.q[1] = *(const uint4*)&As[(r0 + 16) * ASTR + kb + 8];
        int c0 = wx * 32 + lm;
        b0.q[0] = *(const uint4*)&Bs[c0 * BSTR + kb];
        b0.q[1] = *(const uint4*)&Bs[c0 * BSTR + kb + 8];
        b1.q[0] = *(const uint4*)&Bs[(c0 + 16) * BSTR + kb];
        b1.q[1] = *(const uint4*)&Bs[(c0 + 16) * BSTR + kb + 8];

        acc[0][0] = wmma_bf(a0, b0, acc[0][0]);
        acc[0][1] = wmma_bf(a0, b1, acc[0][1]);
        acc[1][0] = wmma_bf(a1, b0, acc[1][0]);
        acc[1][1] = wmma_bf(a1, b1, acc[1][1]);
    }

    // ---- epilogue ----
    #pragma unroll
    for (int i = 0; i < 2; ++i) {
        #pragma unroll
        for (int j = 0; j < 2; ++j) {
            int ncol = bn * BN + wx * 32 + j * 16 + lm;
            int mrow0 = bm * BM + wy * 32 + i * 16 + (lane >> 4) * 8;
            float bb = bias[ncol];
            #pragma unroll
            for (int vv = 0; vv < 8; ++vv) {
                int m = mrow0 + vv;
                float val = acc[i][j][vv] + bb;
                if (MODE == 0) {
                    int nimg = m / 196, p = m % 196;
                    val += pos[(size_t)(1 + p) * DIM + ncol];
                    Xout[((size_t)(nimg * SEQ + 1 + p)) * DIM + ncol] = val;
                } else if (MODE == 1) {
                    if (m < M) {
                        float g = 0.5f * val * (1.f + erff(val * 0.70710678118f));
                        Obf[(size_t)m * MLPD + ncol] = f2bf(g);
                    }
                } else {
                    if (m < M) Xout[(size_t)m * DIM + ncol] += val;
                }
            }
        }
    }
}

// ======================================================================
// LayerNorm: one wave per row of 768, fp32 in -> bf16 out
// ======================================================================
__global__ void __launch_bounds__(256)
k_ln(const float* __restrict__ X, unsigned short* __restrict__ out,
     const float* __restrict__ g, const float* __restrict__ bta)
{
    int lane = threadIdx.x & 31;
    int row = blockIdx.x * 8 + (threadIdx.x >> 5);
    const float4* xr = (const float4*)(X + (size_t)row * DIM);
    float4 vb[6];
    float s = 0.f, s2 = 0.f;
    #pragma unroll
    for (int i = 0; i < 6; ++i) {
        float4 f = xr[i * 32 + lane];
        vb[i] = f;
        s  += f.x + f.y + f.z + f.w;
        s2 += f.x * f.x + f.y * f.y + f.z * f.z + f.w * f.w;
    }
    s = wred_sum(s); s2 = wred_sum(s2);
    float mu  = s * (1.f / DIM);
    float var = s2 * (1.f / DIM) - mu * mu;
    float inv = rsqrtf(var + 1e-5f);
    #pragma unroll
    for (int i = 0; i < 6; ++i) {
        int c = (i * 32 + lane) * 4;
        float4 f = vb[i];
        out[(size_t)row * DIM + c + 0] = f2bf((f.x - mu) * inv * g[c + 0] + bta[c + 0]);
        out[(size_t)row * DIM + c + 1] = f2bf((f.y - mu) * inv * g[c + 1] + bta[c + 1]);
        out[(size_t)row * DIM + c + 2] = f2bf((f.z - mu) * inv * g[c + 2] + bta[c + 2]);
        out[(size_t)row * DIM + c + 3] = f2bf((f.w - mu) * inv * g[c + 3] + bta[c + 3]);
    }
}

// ======================================================================
// QKV: per-head 64x64 projection. grid (50, head, z{q,k,v}), 8 waves/block,
// each wave = 16 rows x 64 cols, K=64 -> 8 WMMA.
// ======================================================================
__global__ void __launch_bounds__(256)
k_qkv(const unsigned short* __restrict__ ln,
      const float* __restrict__ qw, const float* __restrict__ kw, const float* __restrict__ vw,
      const float* __restrict__ qb, const float* __restrict__ kb2, const float* __restrict__ vb,
      unsigned short* __restrict__ qo, unsigned short* __restrict__ ko, unsigned short* __restrict__ vo)
{
    const int h = blockIdx.y;
    const int z = blockIdx.z;
    const float* W  = (z == 0) ? qw : (z == 1) ? kw : vw;
    const float* Bv = (z == 0) ? qb : (z == 1) ? kb2 : vb;
    unsigned short* O = (z == 0) ? qo : (z == 1) ? ko : vo;
    W  += (size_t)h * DH * DH;
    Bv += (size_t)h * DH;

    __shared__ unsigned short Wt[64 * 72];   // transposed: Wt[n][k]
    const int t = threadIdx.x, lane = t & 31, wv = t >> 5;
    {
        int k = t >> 2, n0 = (t & 3) * 16;
        const float4* s4 = (const float4*)(W + (size_t)k * DH + n0);
        #pragma unroll
        for (int i = 0; i < 4; ++i) {
            float4 f = s4[i];
            int nn = n0 + i * 4;
            Wt[(nn + 0) * 72 + k] = f2bf(f.x);
            Wt[(nn + 1) * 72 + k] = f2bf(f.y);
            Wt[(nn + 2) * 72 + k] = f2bf(f.z);
            Wt[(nn + 3) * 72 + k] = f2bf(f.w);
        }
    }
    __syncthreads();

    int m0 = blockIdx.x * 128 + wv * 16;
    if (m0 >= MROWS) return;

    const int lm = lane & 15;
    const int koff = (lane >> 4) * 8;
    const int kb = (lane >> 4) * 16;
    const int m = m0 + lm;

    v8f acc[4];
    #pragma unroll
    for (int i = 0; i < 4; ++i) acc[i] = vzero();

    #pragma unroll
    for (int ks = 0; ks < DH; ks += 32) {
        Frag a;
        const unsigned short* ap = ln + (size_t)m * DIM + h * DH + ks;
        a.q[0] = *(const uint4*)(ap + kb);
        a.q[1] = *(const uint4*)(ap + kb + 8);
        #pragma unroll
        for (int nt = 0; nt < 4; ++nt) {
            Frag b;
            const unsigned short* bp = &Wt[(nt * 16 + lm) * 72 + ks + kb];
            b.q[0] = *(const uint4*)bp;
            b.q[1] = *(const uint4*)(bp + 8);
            acc[nt] = wmma_bf(a, b, acc[nt]);
        }
    }
    // store as [b, h, s, d]
    #pragma unroll
    for (int nt = 0; nt < 4; ++nt) {
        int d = nt * 16 + lm;
        float bb = Bv[d];
        #pragma unroll
        for (int vv = 0; vv < 8; ++vv) {
            int mr = m0 + (lane >> 4) * 8 + vv;
            int nimg = mr / SEQ, s = mr % SEQ;
            O[((size_t)(nimg * NHEAD + h) * SEQ + s) * DH + d] = f2bf(acc[nt][vv] + bb);
        }
    }
}

// ======================================================================
// scores = q @ k^T : one wave per (bh, mtile, ntile), 16x16 out, K=64.
// jobs = 384 * 13 * 13 = 64896 -> 8112 blocks of 8 waves
// ======================================================================
__global__ void __launch_bounds__(256)
k_scores(const unsigned short* __restrict__ q,
         const unsigned short* __restrict__ kk,
         float* __restrict__ sc)
{
    int job  = blockIdx.x * 8 + (threadIdx.x >> 5);
    int lane = threadIdx.x & 31;
    int bh = job / 169, r = job % 169;
    int mt = r / 13, nt = r % 13;
    int lm = lane & 15;
    int koff = (lane >> 4) * 8;
    int kb = (lane >> 4) * 16;
    int i = mt * 16 + lm; int ic = (i > SEQ - 1) ? SEQ - 1 : i;
    int j = nt * 16 + lm; int jc = (j > SEQ - 1) ? SEQ - 1 : j;
    const unsigned short* qp = q  + ((size_t)bh * SEQ + ic) * DH;
    const unsigned short* kp = kk + ((size_t)bh * SEQ + jc) * DH;

    v8f acc = vzero();
    #pragma unroll
    for (int ks = 0; ks < DH; ks += 32) {
        Frag a, b;
        a.q[0] = *(const uint4*)(qp + ks + kb);
        a.q[1] = *(const uint4*)(qp + ks + kb + 8);
        b.q[0] = *(const uint4*)(kp + ks + kb);
        b.q[1] = *(const uint4*)(kp + ks + kb + 8);
        acc = wmma_bf(a, b, acc);
    }
    if (j < SEQ) {
        #pragma unroll
        for (int vv = 0; vv < 8; ++vv) {
            int ii = mt * 16 + (lane >> 4) * 8 + vv;
            if (ii < SEQ) sc[((size_t)bh * SEQ + ii) * SEQ + j] = acc[vv];
        }
    }
}

// ======================================================================
// softmax over 197 logits, scale 1/8, fp32 in -> bf16 out padded to 224
// ======================================================================
__global__ void __launch_bounds__(256)
k_softmax(const float* __restrict__ sc, unsigned short* __restrict__ att)
{
    int lane = threadIdx.x & 31;
    int row = blockIdx.x * 8 + (threadIdx.x >> 5);
    const float* sp = sc + (size_t)row * SEQ;
    float x[7];
    float mx = -3.0e38f;
    #pragma unroll
    for (int i = 0; i < 7; ++i) {
        int j = i * 32 + lane;
        x[i] = (j < SEQ) ? sp[j] * 0.125f : -3.0e38f;
        mx = fmaxf(mx, x[i]);
    }
    mx = wred_max(mx);
    float s = 0.f;
    #pragma unroll
    for (int i = 0; i < 7; ++i) {
        int j = i * 32 + lane;
        float e = (j < SEQ) ? expf(x[i] - mx) : 0.f;
        x[i] = e; s += e;
    }
    s = wred_sum(s);
    float r = 1.f / s;
    unsigned short* op = att + (size_t)row * ATTP;
    #pragma unroll
    for (int i = 0; i < 7; ++i) op[i * 32 + lane] = f2bf(x[i] * r);
}

// ======================================================================
// out = att @ v, accumulated straight into residual X.
// one wave per (bh, mtile, dtile): 16x16 out, K=224 (padded) -> 7 WMMA
// jobs = 384 * 13 * 4 = 19968 -> 2496 blocks
// ======================================================================
__global__ void __launch_bounds__(256)
k_av(const unsigned short* __restrict__ att,
     const unsigned short* __restrict__ vmat,
     float* __restrict__ X)
{
    int job  = blockIdx.x * 8 + (threadIdx.x >> 5);
    int lane = threadIdx.x & 31;
    int bh = job / 52, r = job % 52;
    int mt = r / 4, dt = r % 4;
    int lm = lane & 15;
    int koff = (lane >> 4) * 8;
    int kb = (lane >> 4) * 16;
    int i = mt * 16 + lm; int ic = (i > SEQ - 1) ? SEQ - 1 : i;
    int d = dt * 16 + lm;
    const unsigned short* ap = att + ((size_t)bh * SEQ + ic) * ATTP;
    const unsigned short* vp = vmat + (size_t)bh * SEQ * DH;

    v8f acc = vzero();
    #pragma unroll
    for (int ks = 0; ks < ATTP; ks += 32) {
        Frag a, b;
        a.q[0] = *(const uint4*)(ap + ks + kb);
        a.q[1] = *(const uint4*)(ap + ks + kb + 8);
        #pragma unroll
        for (int e = 0; e < 16; ++e) {
            int j = ks + kb + e;
            if (j > SEQ - 1) j = SEQ - 1;   // att pad cols are 0 -> product 0
            b.u[e] = vp[(size_t)j * DH + d];
        }
        acc = wmma_bf(a, b, acc);
    }
    int n = bh / NHEAD, h = bh % NHEAD;
    #pragma unroll
    for (int vv = 0; vv < 8; ++vv) {
        int ii = mt * 16 + (lane >> 4) * 8 + vv;
        if (ii < SEQ)
            X[((size_t)(n * SEQ) + ii) * DIM + h * DH + d] += acc[vv];
    }
}

// ======================================================================
// host launcher
// ======================================================================
extern "C" void kernel_launch(void* const* d_in, const int* in_sizes, int n_in,
                              void* d_out, int out_size, void* d_ws, size_t ws_size,
                              hipStream_t stream)
{
    (void)in_sizes; (void)n_in; (void)out_size; (void)ws_size;

    const float* images = (const float*)d_in[0];
    const float* map_w  = (const float*)d_in[1];
    const float* map_b  = (const float*)d_in[2];
    const float* cls    = (const float*)d_in[3];
    const float* pos    = (const float*)d_in[4];
    const float* ln1s   = (const float*)d_in[5];
    const float* ln1b   = (const float*)d_in[6];
    const float* qw     = (const float*)d_in[7];
    const float* qb     = (const float*)d_in[8];
    const float* kw     = (const float*)d_in[9];
    const float* kb     = (const float*)d_in[10];
    const float* vw     = (const float*)d_in[11];
    const float* vb     = (const float*)d_in[12];
    const float* ln2s   = (const float*)d_in[13];
    const float* ln2b   = (const float*)d_in[14];
    const float* w1     = (const float*)d_in[15];
    const float* b1     = (const float*)d_in[16];
    const float* w2     = (const float*)d_in[17];
    const float* b2     = (const float*)d_in[18];

    float* X = (float*)d_out;    // residual stream [32,197,768] fp32
    char* ws = (char*)d_ws;

    // workspace layout (bytes, all 256-aligned)
    const size_t LN_B  = (size_t)MROWS * DIM * 2;            //  9,682,944
    const size_t QKV_B = (size_t)BATCH * NHEAD * SEQ * DH * 2; // 9,682,944
    unsigned short* lnb  = (unsigned short*)(ws);
    unsigned short* qbuf = (unsigned short*)(ws + LN_B);
    unsigned short* kbuf = (unsigned short*)(ws + LN_B + QKV_B);
    unsigned short* vbuf = (unsigned short*)(ws + LN_B + 2 * QKV_B);
    unsigned short* h1   = (unsigned short*)(ws + LN_B + 3 * QKV_B);
    const size_t H1_B = (size_t)MROWS * MLPD * 2;            // 38,731,776
    float* sc  = (float*)(ws + LN_B + 3 * QKV_B + H1_B);
    const size_t SC_B = (size_t)BATCH * NHEAD * SEQ * SEQ * 4;
    unsigned short* attb = (unsigned short*)(ws + LN_B + 3 * QKV_B + H1_B + SC_B);

    // row 0 per batch = cls + pos[0]
    k_cls_pos<<<96, 256, 0, stream>>>(X, cls, pos);
    // patchify + embed + pos (M=6272 rows -> X rows 1..196)
    k_gemm<0><<<dim3(49, 12), 256, 0, stream>>>(nullptr, images, map_w, map_b,
                                                pos, X, nullptr, 6272, DIM, DIM);

    for (int l = 0; l < NLAY; ++l) {
        const float* qwl = qw + (size_t)l * NHEAD * DH * DH;
        const float* kwl = kw + (size_t)l * NHEAD * DH * DH;
        const float* vwl = vw + (size_t)l * NHEAD * DH * DH;
        const float* w1l = w1 + (size_t)l * DIM * MLPD;
        const float* w2l = w2 + (size_t)l * MLPD * DIM;

        k_ln<<<788, 256, 0, stream>>>(X, lnb, ln1s + l * DIM, ln1b + l * DIM);
        k_qkv<<<dim3(50, NHEAD, 3), 256, 0, stream>>>(lnb, qwl, kwl, vwl,
                                                      qb + l * DIM, kb + l * DIM, vb + l * DIM,
                                                      qbuf, kbuf, vbuf);
        k_scores<<<8112, 256, 0, stream>>>(qbuf, kbuf, sc);
        k_softmax<<<9456, 256, 0, stream>>>(sc, attb);
        k_av<<<2496, 256, 0, stream>>>(attb, vbuf, X);

        k_ln<<<788, 256, 0, stream>>>(X, lnb, ln2s + l * DIM, ln2b + l * DIM);
        k_gemm<1><<<dim3(50, 48), 256, 0, stream>>>(lnb, nullptr, w1l, b1 + l * MLPD,
                                                    nullptr, nullptr, h1, MROWS, MLPD, DIM);
        k_gemm<2><<<dim3(50, 12), 256, 0, stream>>>(h1, nullptr, w2l, b2 + l * DIM,
                                                    nullptr, X, nullptr, MROWS, DIM, MLPD);
    }
}
